// SpookyNetEmbedding_81277961110082
// MI455X (gfx1250) — matrix operand order repaired and checked by reading, not compiled, not run on hardware
//
#include <hip/hip_runtime.h>
#include <hip/hip_bf16.h>

// ---------------------------------------------------------------------------
// CDNA5 (gfx1250) SpookyNet forward.  All dense 128-K matmuls run through
// v_wmma_f32_16x16x32_f16 (f16 in / f32 accumulate).  wave32 layouts per
// CDNA5 ISA.  Fast-math: v_rcp_f32 instead of IEEE divide chains.
// ---------------------------------------------------------------------------

typedef __attribute__((ext_vector_type(16))) _Float16 v16h;
typedef __attribute__((ext_vector_type(8)))  _Float16 v8h;
typedef __attribute__((ext_vector_type(8)))  float    v8f;

__device__ __constant__ int c_comb15[16] = {1,15,105,455,1365,3003,5005,6435,
                                            6435,5005,3003,1365,455,105,15,1};

__device__ __forceinline__ float fast_rcp(float x) {
  return __builtin_amdgcn_rcpf(x);          // single v_rcp_f32
}

__device__ __forceinline__ float dev_silu(float x) {
  return x * fast_rcp(1.0f + __expf(-x));   // v_exp + v_add + v_rcp + v_mul
}

__device__ __forceinline__ v8f wmma_f16(v16h a, v16h b, v8f c) {
  // (neg_a, A, neg_b, B, c_mod, C, reuse_a, reuse_b)
  return __builtin_amdgcn_wmma_f32_16x16x32_f16(false, a, false, b,
                                                (short)0, c, false, false);
}

// A fragment: f32 source, optional SiLU, convert to f16.
// lane<16 (hs=0): row = ln, K = {k0..k0+15}
// lane>=16 (hs=1): row = ln, K = {k0+16..k0+31}
template<bool ACT>
__device__ __forceinline__ v16h load_a32(const float* __restrict__ A,
                                         size_t rowoff, int k0, int hs) {
  const float* p = A + rowoff + k0 + hs * 16;
  float4 f0 = *(const float4*)(p);
  float4 f1 = *(const float4*)(p + 4);
  float4 f2 = *(const float4*)(p + 8);
  float4 f3 = *(const float4*)(p + 12);
  float t[16] = {f0.x,f0.y,f0.z,f0.w, f1.x,f1.y,f1.z,f1.w,
                 f2.x,f2.y,f2.z,f2.w, f3.x,f3.y,f3.z,f3.w};
  v16h r;
#pragma unroll
  for (int i = 0; i < 16; ++i) {
    float v = t[i];
    if (ACT) v = dev_silu(v);
    r[i] = (_Float16)v;
  }
  return r;
}

// ---------------------------------------------------------------------------
// Generic GEMM: Y[M x Ncols] = act(A[M x 128]) @ W[128 x Ncols] (+bias)(+resid)
// W is pre-transposed f16: BT[n][128].  8 waves/block, 128 rows x 128 cols.
// ---------------------------------------------------------------------------
template<bool ACT>
__global__ __launch_bounds__(256) void k_gemm128(
    const float* __restrict__ A, const _Float16* __restrict__ BT,
    const float* __restrict__ bias, const float* __restrict__ resid,
    float* __restrict__ Y, int M, int Ncols)
{
  const int wave = threadIdx.x >> 5;
  const int lane = threadIdx.x & 31;
  const int hs   = lane >> 4;
  const int ln   = lane & 15;
  const int row0 = blockIdx.x * 128 + wave * 16;
  if (row0 >= M) return;                       // wave-uniform
  const int colb = blockIdx.y * 128;

  v8f acc[8];
#pragma unroll
  for (int t = 0; t < 8; ++t)
#pragma unroll
    for (int v = 0; v < 8; ++v) acc[t][v] = 0.0f;

  const size_t rowoff = (size_t)(row0 + ln) * 128;
#pragma unroll
  for (int k0 = 0; k0 < 128; k0 += 32) {
    v16h a = load_a32<ACT>(A, rowoff, k0, hs);
#pragma unroll
    for (int t = 0; t < 8; ++t) {
      const _Float16* bp = BT + (size_t)(colb + t * 16 + ln) * 128 + k0 + hs * 16;
      v16h b = *(const v16h*)bp;               // lanes0-15: K=k0..15, 16-31: +16
      acc[t] = wmma_f16(a, b, acc[t]);
    }
  }
#pragma unroll
  for (int t = 0; t < 8; ++t) {
    const int coln = colb + t * 16 + ln;
    const float bv = bias ? bias[coln] : 0.0f;
#pragma unroll
    for (int v = 0; v < 8; ++v) {
      const int r = row0 + v + hs * 8;         // C/D layout: M = v + 8*hs
      const size_t idx = (size_t)r * Ncols + coln;
      float val = acc[t][v] + bv;
      if (resid) val += resid[idx];
      Y[idx] = val;
    }
  }
}

// ---------------------------------------------------------------------------
// Edge GEMM + segment-sum scatter:
//   out[edge_src[e]*RPE + c][n] += feat[edge_dst[e]][n] * (basis[row] @ WG)[n]
// basis rows are 16 halves (K=16, zero-padded to 32 in the fragments).
// ---------------------------------------------------------------------------
template<int RPE>
__global__ __launch_bounds__(256) void k_gemm_scatter(
    const _Float16* __restrict__ basisA, const _Float16* __restrict__ BT,
    const float* __restrict__ feat, const int* __restrict__ esrc,
    const int* __restrict__ edst, float* __restrict__ out, int M)
{
  const int wave = threadIdx.x >> 5;
  const int lane = threadIdx.x & 31;
  const int hs   = lane >> 4;
  const int ln   = lane & 15;
  const int row0 = blockIdx.x * 128 + wave * 16;
  if (row0 >= M) return;

  v8f acc[8];
#pragma unroll
  for (int t = 0; t < 8; ++t)
#pragma unroll
    for (int v = 0; v < 8; ++v) acc[t][v] = 0.0f;

  // A fragment: K<16 valid halves, rest zero.
  v16h a;
#pragma unroll
  for (int i = 0; i < 16; ++i) a[i] = (_Float16)0.0f;
  {
    const v8h lo = *(const v8h*)(basisA + (size_t)(row0 + ln) * 16 + hs * 8);
#pragma unroll
    for (int i = 0; i < 8; ++i) a[i] = lo[i];
  }
#pragma unroll
  for (int t = 0; t < 8; ++t) {
    v16h b;
#pragma unroll
    for (int i = 0; i < 16; ++i) b[i] = (_Float16)0.0f;
    {
      const v8h bl = *(const v8h*)(BT + (size_t)(t * 16 + ln) * 16 + hs * 8); // K=8*hs..8*hs+7
#pragma unroll
      for (int i = 0; i < 8; ++i) b[i] = bl[i];
    }
    acc[t] = wmma_f16(a, b, acc[t]);
  }
#pragma unroll
  for (int v = 0; v < 8; ++v) {
    const int row = row0 + v + hs * 8;
    const int e   = row / RPE;
    const int cc  = row - e * RPE;
    const int s   = esrc[e];
    const int d   = edst[e];
    const float* frow = feat + (size_t)d * 128;
    float* orow = out + ((size_t)s * RPE + cc) * 128;
#pragma unroll
    for (int t = 0; t < 8; ++t) {
      const int coln = t * 16 + ln;
      atomicAdd(&orow[coln], frow[coln] * acc[t][v]);
    }
  }
}

// ---------------------------------------------------------------------------
// Weight prep: WT[n][K] (f16) = W[kk][n] (f32)
// ---------------------------------------------------------------------------
__global__ void k_prep(const float* __restrict__ W, _Float16* __restrict__ WT,
                       int K, int N)
{
  int idx = blockIdx.x * blockDim.x + threadIdx.x;
  if (idx >= K * N) return;
  int n = idx / K, kk = idx - n * K;
  WT[idx] = (_Float16)W[(size_t)kk * N + n];
}

// eZ = enc_table[species] @ W_species + rand_table[species]
__global__ void k_embed(const int* __restrict__ species,
                        const float* __restrict__ enc,
                        const float* __restrict__ Wsp,
                        const float* __restrict__ randt,
                        float* __restrict__ eZ, int NA)
{
  int idx = blockIdx.x * blockDim.x + threadIdx.x;
  if (idx >= NA * 128) return;
  int a = idx >> 7, n = idx & 127;
  int s = species[a];
  float acc = randt[(size_t)s * 128 + n];
  const float* er = enc + (size_t)s * 20;
#pragma unroll
  for (int d = 0; d < 20; ++d) acc += er[d] * Wsp[(size_t)d * 128 + n];
  eZ[idx] = acc;
}

// attention pass1: one wave32 per atom, dot(qi, k) -> softplus -> wnorm
__global__ void k_attn1(const float* __restrict__ qi,
                        const float* __restrict__ kvtab, int sign_select,
                        const float* __restrict__ tot, const int* __restrict__ bidx,
                        float* __restrict__ wi, float* __restrict__ wnorm, int NA)
{
  int a = (blockIdx.x * blockDim.x + threadIdx.x) >> 5;
  int lane = threadIdx.x & 31;
  if (a >= NA) return;
  int b = bidx[a];
  const float* krow = kvtab + ((sign_select && tot[b] < 0.0f) ? 128 : 0);
  float s = 0.0f;
  for (int i = lane; i < 128; i += 32) s += qi[(size_t)a * 128 + i] * krow[i];
#pragma unroll
  for (int off = 16; off; off >>= 1) s += __shfl_xor(s, off, 32);
  if (lane == 0) {
    float x = s * 0.08838834764831845f;       // 1/sqrt(128)
    float w = (x > 20.0f) ? x : log1pf(__expf(x));
    wi[a] = w;
    atomicAdd(&wnorm[b], w);
  }
}

// attention pass2: avi = wi * (tot/wnorm) * v_row
__global__ void k_attn2(const float* __restrict__ wi, const float* __restrict__ wnorm,
                        const float* __restrict__ tot, const int* __restrict__ bidx,
                        const float* __restrict__ vpos, const float* __restrict__ vneg,
                        float* __restrict__ avi, int NA)
{
  int idx = blockIdx.x * blockDim.x + threadIdx.x;
  if (idx >= NA * 128) return;
  int a = idx >> 7, n = idx & 127;
  int b = bidx[a];
  const float* v = (tot[b] >= 0.0f) ? vpos : vneg;
  avi[idx] = wi[a] * (tot[b] * fast_rcp(wnorm[b])) * v[n];
}

// edge basis: Bernstein radial (16) x sph-harm channels -> f16 rows of 16
__global__ void k_edge_basis(const float* __restrict__ vec,
                             const float* __restrict__ sw,
                             const float* __restrict__ gamma_p,
                             _Float16* __restrict__ bs, _Float16* __restrict__ bp,
                             _Float16* __restrict__ bd, int E)
{
  int e = blockIdx.x * blockDim.x + threadIdx.x;
  if (e >= E) return;
  float vx = vec[3 * e], vy = vec[3 * e + 1], vz = vec[3 * e + 2];
  float dist = sqrtf(vx * vx + vy * vy + vz * vz);
  dist = fmaxf(dist, 1e-6f);
  float inv = fast_rcp(dist);
  float x = vx * inv, y = vy * inv, z = vz * inv;
  const float s3 = 1.7320508075688772f;
  float Y[9] = {1.0f, x, y, z, s3 * x * y, s3 * y * z,
                1.5f * z * z - 0.5f, s3 * x * z, 0.5f * s3 * (x * x - y * y)};
  float g = gamma_p[0];
  float sp = (g > 20.0f) ? g : log1pf(__expf(g));
  float f = __expf(-sp * dist);
  const float eps = 1e-10f;
  float lf = __logf(f + eps), lomf = __logf(1.0f - f + eps);
  float swv = sw[e];
  float rb[16];
#pragma unroll
  for (int k = 0; k < 16; ++k)
    rb[k] = __expf(__logf((float)c_comb15[k]) + k * lf + (15 - k) * lomf) * swv;
#pragma unroll
  for (int k = 0; k < 16; ++k) bs[(size_t)e * 16 + k] = (_Float16)rb[k];
#pragma unroll
  for (int c = 0; c < 3; ++c)
#pragma unroll
    for (int k = 0; k < 16; ++k)
      bp[((size_t)e * 3 + c) * 16 + k] = (_Float16)(rb[k] * Y[1 + c]);
#pragma unroll
  for (int c = 0; c < 5; ++c)
#pragma unroll
    for (int k = 0; k < 16; ++k)
      bd[((size_t)e * 5 + c) * 16 + k] = (_Float16)(rb[k] * Y[4 + c]);
}

// P12[a][d] = sum_c PP[(a*RPE+c)][d] * PP[(a*RPE+c)][128+d]   (PP rows of 256)
template<int RPE>
__global__ void k_pdot(const float* __restrict__ PP, float* __restrict__ out, int NA)
{
  int idx = blockIdx.x * blockDim.x + threadIdx.x;
  if (idx >= NA * 128) return;
  int a = idx >> 7, d = idx & 127;
  float s = 0.0f;
#pragma unroll
  for (int c = 0; c < RPE; ++c) {
    const float* row = PP + ((size_t)a * RPE + c) * 256;
    s += row[d] * row[128 + d];
  }
  out[idx] = s;
}

__global__ void k_add4(const float* __restrict__ a, const float* __restrict__ b,
                       const float* __restrict__ c, const float* __restrict__ d,
                       float* __restrict__ o, int n)
{
  int i = blockIdx.x * blockDim.x + threadIdx.x;
  if (i >= n) return;
  float v = a[i] + b[i];
  if (c) v += c[i];
  if (d) v += d[i];
  o[i] = v;
}

// ---------------------------------------------------------------------------
// Host orchestration
// ---------------------------------------------------------------------------
static inline int hcdiv(long a, long b) { return (int)((a + b - 1) / b); }

struct PrepItem { const float* src; _Float16* dst; int K; int N; };
struct ResW { _Float16 *W1, *W2, *Wo; const float *b1, *b2, *bo; };

extern "C" void kernel_launch(void* const* d_in, const int* in_sizes, int n_in,
                              void* d_out, int out_size, void* d_ws, size_t ws_size,
                              hipStream_t stream)
{
  (void)out_size;
  if (n_in < 187) return;
  const int NA = in_sizes[0];
  const int E  = in_sizes[1];
  const int NB = in_sizes[6];

  const int*   species = (const int*)d_in[0];
  const int*   esrc    = (const int*)d_in[1];
  const int*   edst    = (const int*)d_in[2];
  const float* vec     = (const float*)d_in[3];
  const float* sw      = (const float*)d_in[4];
  const int*   bidx    = (const int*)d_in[5];
  const float* Qtot    = (const float*)d_in[6];
  const float* Stot    = (const float*)d_in[7];

  // jax pytree (sorted-key) flattening of params:
  const int I_Wsp = 8, I_Wq = 9, I_Ws = 10, I_bq = 11, I_bs = 12,
            I_eQ = 13, I_eS = 18, I_enc = 23, I_gamma = 24,
            I_kvc = 25, I_kvs = 26, I_L0 = 27, LSTR = 53;
  const int I_rand = I_L0 + 3 * LSTR;  // 186
  // per-layer leaf offsets (sorted): WD=0 WGd=1 WGp=2 WGs=3 WP=4
  //   c=5 d=11 l=17 p=23 s=29 xi=35 xtilde=41 y=47  (resmlp leaves W1,W2,Wo,b1,b2,bo)

  char* wsbase = (char*)d_ws;
  size_t off = 0;
  auto alloc = [&](size_t bytes) -> void* {
    size_t o = (off + 255) & ~(size_t)255;
    off = o + bytes;
    return wsbase + o;
  };

  PrepItem items[128]; int ni = 0;
  auto allocW = [&](int leaf, int K, int N) -> _Float16* {
    _Float16* p = (_Float16*)alloc((size_t)K * N * sizeof(_Float16));
    items[ni++] = { (const float*)d_in[leaf], p, K, N };
    return p;
  };
  auto resW = [&](int leaf, bool has_bo) -> ResW {
    ResW r;
    r.W1 = allocW(leaf + 0, 128, 128);
    r.W2 = allocW(leaf + 1, 128, 128);
    r.Wo = allocW(leaf + 2, 128, 128);
    r.b1 = (const float*)d_in[leaf + 3];
    r.b2 = (const float*)d_in[leaf + 4];
    r.bo = has_bo ? (const float*)d_in[leaf + 5] : nullptr;
    return r;
  };

  _Float16* WqT = allocW(I_Wq, 128, 128);
  _Float16* WsT = allocW(I_Ws, 128, 128);
  ResW eQw = resW(I_eQ, false);
  ResW eSw = resW(I_eS, false);
  _Float16 *WGsT[3], *WGpT[3], *WGdT[3], *WPT[3], *WDT[3];
  ResW Lc[3], Ldr[3], Ll[3], Lpr[3], Lsr[3], Lxi[3], Lxt[3], Ly[3];
  for (int li = 0; li < 3; ++li) {
    int b0 = I_L0 + li * LSTR;
    WDT[li]  = allocW(b0 + 0, 128, 256);
    WGdT[li] = allocW(b0 + 1, 16, 128);
    WGpT[li] = allocW(b0 + 2, 16, 128);
    WGsT[li] = allocW(b0 + 3, 16, 128);
    WPT[li]  = allocW(b0 + 4, 128, 256);
    Lc[li]  = resW(b0 + 5, true);
    Ldr[li] = resW(b0 + 11, true);
    Ll[li]  = resW(b0 + 17, true);
    Lpr[li] = resW(b0 + 23, true);
    Lsr[li] = resW(b0 + 29, true);
    Lxi[li] = resW(b0 + 35, true);
    Lxt[li] = resW(b0 + 41, true);
    Ly[li]  = resW(b0 + 47, true);
  }

  const size_t NF = (size_t)NA * 128;
  float* eZ    = (float*)alloc(NF * 4);
  float* xi    = (float*)alloc(NF * 4);
  float* t0    = (float*)alloc(NF * 4);
  float* t1    = (float*)alloc(NF * 4);
  float* avi   = (float*)alloc(NF * 4);
  float* xtl   = (float*)alloc(NF * 4);
  float* cbuf  = (float*)alloc(NF * 4);
  float* sjb   = (float*)alloc(NF * 4);
  float* pjb   = (float*)alloc(NF * 4);
  float* djb   = (float*)alloc(NF * 4);
  float* lbuf  = (float*)alloc(NF * 4);
  float* sumb  = (float*)alloc(NF * 4);
  float* P12   = (float*)alloc(NF * 4);
  float* D12   = (float*)alloc(NF * 4);
  float* si    = (float*)alloc(NF * 4);
  float* pi    = (float*)alloc(NF * 4 * 3);
  float* di    = (float*)alloc(NF * 4 * 5);
  float* PP    = (float*)alloc((size_t)NA * 3 * 256 * 4);
  float* DD    = (float*)alloc((size_t)NA * 5 * 256 * 4);
  float* wib   = (float*)alloc((size_t)NA * 4);
  float* wnorm = (float*)alloc((size_t)NB * 4);
  _Float16* bs16 = (_Float16*)alloc((size_t)E * 16 * 2);
  _Float16* bp16 = (_Float16*)alloc((size_t)E * 3 * 16 * 2);
  _Float16* bd16 = (_Float16*)alloc((size_t)E * 5 * 16 * 2);
  if (off > ws_size) return;  // insufficient workspace

  float* yout  = (float*)d_out;
  float* eQout = yout + NF;
  float* eSout = yout + 2 * NF;

  // weight prep (f32 -> transposed f16, rebuilt every call: deterministic)
  for (int i = 0; i < ni; ++i) {
    int tot = items[i].K * items[i].N;
    k_prep<<<hcdiv(tot, 256), 256, 0, stream>>>(items[i].src, items[i].dst,
                                                items[i].K, items[i].N);
  }

  auto gemm = [&](bool act, const float* A, const _Float16* BT, const float* bias,
                  const float* resid, float* Y, int M, int Ncols) {
    dim3 g(hcdiv(M, 128), Ncols / 128);
    if (act) k_gemm128<true><<<g, 256, 0, stream>>>(A, BT, bias, resid, Y, M, Ncols);
    else     k_gemm128<false><<<g, 256, 0, stream>>>(A, BT, bias, resid, Y, M, Ncols);
  };
  auto resmlp = [&](const ResW& w, const float* x, const float* resid_out,
                    float* out, bool res_only) {
    gemm(true, x, w.W1, w.b1, nullptr, t0, NA, 128);                 // h
    gemm(true, t0, w.W2, w.b2, x, res_only ? out : t1, NA, 128);     // r = x+h2
    if (!res_only) gemm(true, t1, w.Wo, w.bo, resid_out, out, NA, 128);
  };

  // ---- embedding ----
  k_embed<<<hcdiv((long)NF, 256), 256, 0, stream>>>(
      species, (const float*)d_in[I_enc], (const float*)d_in[I_Wsp],
      (const float*)d_in[I_rand], eZ, NA);

  // ---- charge attention ----
  gemm(false, eZ, WqT, (const float*)d_in[I_bq], nullptr, t0, NA, 128);   // qi
  hipMemsetAsync(wnorm, 0, (size_t)NB * 4, stream);
  k_attn1<<<hcdiv(NA, 8), 256, 0, stream>>>(t0, (const float*)d_in[I_kvc], 1,
                                            Qtot, bidx, wib, wnorm, NA);
  k_attn2<<<hcdiv((long)NF, 256), 256, 0, stream>>>(
      wib, wnorm, Qtot, bidx, (const float*)d_in[I_kvc] + 2 * 128,
      (const float*)d_in[I_kvc] + 3 * 128, avi, NA);
  resmlp(eQw, avi, nullptr, eQout, false);
  k_add4<<<hcdiv((long)NF, 256), 256, 0, stream>>>(eZ, eQout, nullptr, nullptr,
                                                   xi, (int)NF);

  // ---- spin attention ----
  gemm(false, eZ, WsT, (const float*)d_in[I_bs], nullptr, t0, NA, 128);   // si_
  hipMemsetAsync(wnorm, 0, (size_t)NB * 4, stream);
  k_attn1<<<hcdiv(NA, 8), 256, 0, stream>>>(t0, (const float*)d_in[I_kvs], 0,
                                            Stot, bidx, wib, wnorm, NA);
  k_attn2<<<hcdiv((long)NF, 256), 256, 0, stream>>>(
      wib, wnorm, Stot, bidx, (const float*)d_in[I_kvs] + 128,
      (const float*)d_in[I_kvs] + 128, avi, NA);
  resmlp(eSw, avi, nullptr, eSout, false);
  k_add4<<<hcdiv((long)NF, 256), 256, 0, stream>>>(xi, eSout, nullptr, nullptr,
                                                   xi, (int)NF);

  // ---- edge basis (layer-invariant) ----
  k_edge_basis<<<hcdiv(E, 256), 256, 0, stream>>>(
      vec, sw, (const float*)d_in[I_gamma], bs16, bp16, bd16, E);

  // ---- message-passing layers ----
  hipMemsetAsync(yout, 0, NF * 4, stream);
  for (int li = 0; li < 3; ++li) {
    resmlp(Lxt[li], xi, nullptr, xtl, true);        // xtilde (res-only)
    resmlp(Lc[li],  xtl, nullptr, cbuf, false);
    resmlp(Lsr[li], xtl, nullptr, sjb,  false);
    resmlp(Lpr[li], xtl, nullptr, pjb,  false);
    resmlp(Ldr[li], xtl, nullptr, djb,  false);

    hipMemsetAsync(si, 0, NF * 4, stream);
    hipMemsetAsync(pi, 0, NF * 4 * 3, stream);
    hipMemsetAsync(di, 0, NF * 4 * 5, stream);
    k_gemm_scatter<1><<<hcdiv(E, 128), 256, 0, stream>>>(
        bs16, WGsT[li], sjb, esrc, edst, si, E);
    k_gemm_scatter<3><<<hcdiv(3L * E, 128), 256, 0, stream>>>(
        bp16, WGpT[li], pjb, esrc, edst, pi, 3 * E);
    k_gemm_scatter<5><<<hcdiv(5L * E, 128), 256, 0, stream>>>(
        bd16, WGdT[li], djb, esrc, edst, di, 5 * E);

    gemm(false, pi, WPT[li], nullptr, nullptr, PP, NA * 3, 256);
    gemm(false, di, WDT[li], nullptr, nullptr, DD, NA * 5, 256);
    k_pdot<3><<<hcdiv((long)NF, 256), 256, 0, stream>>>(PP, P12, NA);
    k_pdot<5><<<hcdiv((long)NF, 256), 256, 0, stream>>>(DD, D12, NA);

    k_add4<<<hcdiv((long)NF, 256), 256, 0, stream>>>(cbuf, si, P12, D12, sumb, (int)NF);
    resmlp(Ll[li], sumb, nullptr, lbuf, false);
    k_add4<<<hcdiv((long)NF, 256), 256, 0, stream>>>(xtl, lbuf, nullptr, nullptr,
                                                     sumb, (int)NF);
    resmlp(Lxi[li], sumb, nullptr, xi, false);
    resmlp(Ly[li],  xi, yout, yout, false);         // y += y_mlp(xi)
  }
}
